// MultiHeadSelfAttention_27487790694990
// MI455X (gfx1250) — matrix operand order repaired and checked by reading, not compiled, not run on hardware
//
#include <hip/hip_runtime.h>

// ---------------------------------------------------------------------------
// MI455X (gfx1250) fused multi-head self-attention block.
// LN -> QKV gemm (f16 WMMA) -> flash attention (f16 WMMA, TDM-staged K/V)
//    -> out proj + residual
// ---------------------------------------------------------------------------

#define BATCH 4
#define SEQ   2048
#define CH    1024
#define HEADS 16
#define HDIM  64
#define ROWS  (BATCH * SEQ)          // 8192
#define C3    (3 * CH)               // 3072

typedef __attribute__((ext_vector_type(16))) _Float16 v16h;
typedef __attribute__((ext_vector_type(8)))  _Float16 half8;
typedef __attribute__((ext_vector_type(8)))  float    v8f;
typedef __attribute__((ext_vector_type(4)))  unsigned int v4u;
typedef __attribute__((ext_vector_type(4)))  int      v4i;
typedef __attribute__((ext_vector_type(8)))  int      v8i;

// ---- WMMA wrapper: D = A(16x32 f16) * B(32x16 f16) + C(16x16 f32) ----------
__device__ __forceinline__ v8f wmma_f16(v16h a, v16h b, v8f c) {
  return __builtin_amdgcn_wmma_f32_16x16x32_f16(
      /*neg_a=*/false, a, /*neg_b=*/false, b,
      /*c_mod=*/(short)0, c, /*reuse_a=*/false, /*reuse_b=*/false);
}

// A-fragment (16x32, 16-bit): lane holds row = idx; halves e<8 -> k=e+8*hi,
// e>=8 -> k=e+8+8*hi. Source row-major with leading dim ld (elements).
__device__ __forceinline__ v16h load_a_frag(const _Float16* base, int ld,
                                            int row, int kb, int hi) {
  const _Float16* p = base + row * ld + kb + hi * 8;
  half8 lo = *(const half8*)p;
  half8 up = *(const half8*)(p + 16);
  v16h r;
#pragma unroll
  for (int i = 0; i < 8; ++i) { r[i] = lo[i]; r[i + 8] = up[i]; }
  return r;
}

// B-fragment (32x16, 16-bit): lane holds column n = idx; e<8 -> k=e+8*hi,
// e>=8 -> k=e+8+8*hi.
// Source is Bt[n][k] row-major (n-major, contiguous k) with leading dim ld.
__device__ __forceinline__ v16h load_b_frag(const _Float16* base, int ld,
                                            int row, int kb, int hi) {
  const _Float16* p = base + row * ld + kb + hi * 8;
  half8 lo = *(const half8*)p;
  half8 up = *(const half8*)(p + 16);
  v16h r;
#pragma unroll
  for (int i = 0; i < 8; ++i) { r[i] = lo[i]; r[i + 8] = up[i]; }
  return r;
}

// Reductions inside each 16-lane half of a wave32 (keeps hi/lo halves apart).
__device__ __forceinline__ float rmax16(float v) {
  v = fmaxf(v, __shfl_xor(v, 1, 32));
  v = fmaxf(v, __shfl_xor(v, 2, 32));
  v = fmaxf(v, __shfl_xor(v, 4, 32));
  v = fmaxf(v, __shfl_xor(v, 8, 32));
  return v;
}
__device__ __forceinline__ float rsum16(float v) {
  v += __shfl_xor(v, 1, 32);
  v += __shfl_xor(v, 2, 32);
  v += __shfl_xor(v, 4, 32);
  v += __shfl_xor(v, 8, 32);
  return v;
}

// ---------------------------------------------------------------------------
// Tensor Data Mover: async 2D tile load global -> LDS (CDNA5 ISA).
// D# group0: [1:0]=count=1, [63:32]=lds byte addr, [120:64]=global addr,
//            [127:126]=type=2.  group1: data_size=2B, pad_enable, pad codes,
//            tensor_dim0/1, tile_dim0/1, tensor_dim0_stride.
// pad_interval code: DWORDs before pad = 2<<code; pad_amount code: DWORDs-1.
// ---------------------------------------------------------------------------
__device__ __forceinline__ void tdm_load_2d(unsigned lds_off, const void* gptr,
                                            unsigned ten_d0, unsigned ten_d1,
                                            unsigned tile_d0, unsigned tile_d1,
                                            unsigned stride0,
                                            unsigned pad_int, unsigned pad_amt) {
  unsigned long long ga = (unsigned long long)(size_t)gptr;
  v4u g0;
  g0[0] = 1u;                                            // count=1 (valid, user)
  g0[1] = lds_off;                                       // LDS byte address
  g0[2] = (unsigned)(ga & 0xffffffffu);                  // global addr lo
  g0[3] = (unsigned)((ga >> 32) & 0x01ffffffu) | (2u << 30);  // addr hi | type=2
  v8i g1;
  g1[0] = (int)((1u << 16) |                             // data_size = 2 bytes
                (1u << 20) |                             // pad_enable
                (pad_int << 22) | (pad_amt << 25));
  g1[1] = (int)((ten_d0 & 0xffffu) << 16);               // tensor_dim0 lo16
  g1[2] = (int)(((ten_d0 >> 16) & 0xffffu) | ((ten_d1 & 0xffffu) << 16));
  g1[3] = (int)(((ten_d1 >> 16) & 0xffffu) | ((tile_d0 & 0xffffu) << 16));
  g1[4] = (int)(tile_d1 & 0xffffu);                      // tile_dim1, tile_dim2=0
  g1[5] = (int)stride0;                                  // tensor_dim0_stride lo
  g1[6] = 0;
  g1[7] = 0;
  v4i z4 = {0, 0, 0, 0};
#if __clang_major__ >= 23
  v8i z8 = {0, 0, 0, 0, 0, 0, 0, 0};
  __builtin_amdgcn_tensor_load_to_lds(g0, g1, z4, z4, z8, 0);
#else
  __builtin_amdgcn_tensor_load_to_lds(g0, g1, z4, z4, 0);
#endif
}

// ---------------------------------------------------------------------------
// 1) LayerNorm (f32 in) -> f16 activations
// ---------------------------------------------------------------------------
__global__ void __launch_bounds__(256) ln_kernel(const float* __restrict__ x,
                                                 const float* __restrict__ g,
                                                 const float* __restrict__ b,
                                                 _Float16* __restrict__ h) {
  const int row = blockIdx.x;
  const float* xr = x + (size_t)row * CH;
  float s = 0.f, ss = 0.f;
  for (int c = threadIdx.x; c < CH; c += 256) {
    float v = xr[c];
    s += v; ss += v * v;
  }
  __shared__ float red[256], red2[256];
  red[threadIdx.x] = s; red2[threadIdx.x] = ss;
  __syncthreads();
  for (int off = 128; off > 0; off >>= 1) {
    if (threadIdx.x < off) {
      red[threadIdx.x]  += red[threadIdx.x + off];
      red2[threadIdx.x] += red2[threadIdx.x + off];
    }
    __syncthreads();
  }
  const float mu  = red[0] * (1.0f / CH);
  const float var = red2[0] * (1.0f / CH) - mu * mu;
  const float rs  = rsqrtf(var + 1e-5f);
  for (int c = threadIdx.x; c < CH; c += 256) {
    float v = (xr[c] - mu) * rs * g[c] + b[c];
    h[(size_t)row * CH + c] = (_Float16)v;
  }
}

// ---------------------------------------------------------------------------
// 2) Weight transpose + f32->f16:  Wt[n*K + k] = W[k*N + n]
// ---------------------------------------------------------------------------
__global__ void __launch_bounds__(256) wtrans_kernel(const float* __restrict__ w,
                                                     _Float16* __restrict__ wt,
                                                     int K, int N) {
  int i = blockIdx.x * 256 + threadIdx.x;
  if (i >= K * N) return;
  int n = i / K, k = i - n * K;
  wt[i] = (_Float16)w[(size_t)k * N + n];
}

// ---------------------------------------------------------------------------
// 3) QKV GEMM:  [8192 x 1024] f16  @  Wt[3072][1024] f16  (+bias)
//    64x64 block tile, 4 waves, K-step 32, LDS staged + next-tile prefetch.
//    Epilogue scatters Q,K as [B,H,T,64] and V transposed as [B,H,64,T].
// ---------------------------------------------------------------------------
__global__ void __launch_bounds__(128) qkv_gemm(const _Float16* __restrict__ A,
                                                const _Float16* __restrict__ Wt,
                                                const float* __restrict__ bias,
                                                _Float16* __restrict__ Qo,
                                                _Float16* __restrict__ Ko,
                                                _Float16* __restrict__ Vto) {
  __shared__ __align__(16) _Float16 at[64 * 40];
  __shared__ __align__(16) _Float16 bt[64 * 40];
  const int tid = threadIdx.x;
  const int w = tid >> 5, lane = tid & 31, idx = lane & 15, hi = lane >> 4;
  const int n0 = blockIdx.x * 64, m0 = blockIdx.y * 64;
  v8f acc[4] = {};
  for (int k0 = 0; k0 < CH; k0 += 32) {
#pragma unroll
    for (int i = tid; i < 256; i += 128) {           // 64x32 halves, 8 per thread
      int flat = i * 8, r = flat >> 5, c = flat & 31;
      *(half8*)&at[r * 40 + c] = *(const half8*)&A [(size_t)(m0 + r) * CH + k0 + c];
      *(half8*)&bt[r * 40 + c] = *(const half8*)&Wt[(size_t)(n0 + r) * CH + k0 + c];
      if (k0 + 32 < CH) {                            // warm L2/L0 for next step
        __builtin_prefetch(&A [(size_t)(m0 + r) * CH + k0 + 32 + c], 0, 3);
        __builtin_prefetch(&Wt[(size_t)(n0 + r) * CH + k0 + 32 + c], 0, 3);
      }
    }
    __syncthreads();
    v16h af = load_a_frag(at, 40, w * 16 + idx, 0, hi);
#pragma unroll
    for (int t = 0; t < 4; ++t)
      acc[t] = wmma_f16(af, load_b_frag(bt, 40, t * 16 + idx, 0, hi), acc[t]);
    __syncthreads();
  }
#pragma unroll
  for (int t = 0; t < 4; ++t) {
    const int gcol = n0 + t * 16 + idx;
    const float bv = bias[gcol];
    const int sec = gcol >> 10, rem = gcol & 1023, head = rem >> 6, d = rem & 63;
#pragma unroll
    for (int r = 0; r < 8; ++r) {
      const int grow = m0 + w * 16 + r + 8 * hi;     // global token row
      const int bb = grow >> 11, tok = grow & 2047;
      const _Float16 v = (_Float16)(acc[t][r] + bv);
      const size_t bhead = (size_t)(bb * HEADS + head);
      if (sec == 0)      Qo [(bhead * SEQ + tok) * HDIM + d] = v;
      else if (sec == 1) Ko [(bhead * SEQ + tok) * HDIM + d] = v;
      else               Vto[(bhead * HDIM + d) * SEQ + tok] = v;
    }
  }
}

// ---------------------------------------------------------------------------
// 4) Flash attention: 1 wave = 16 query rows; block = 4 waves sharing
//    TDM-staged 32-key K chunks + Vt chunks.  scale = 1/sqrt(64) = 0.125.
// ---------------------------------------------------------------------------
__global__ void __launch_bounds__(128) attn_kernel(const _Float16* __restrict__ Q,
                                                   const _Float16* __restrict__ Kt,
                                                   const _Float16* __restrict__ Vt,
                                                   _Float16* __restrict__ O) {
  __shared__ __align__(16) _Float16 kbuf[32 * 72];      // [key][64 + 8 pad]
  __shared__ __align__(16) _Float16 vtbuf[64 * 40];     // [d][32 + 8 pad]
  __shared__ __align__(16) _Float16 pbuf[4 * 16 * 40];  // per-wave P re-stripe
  const int tid = threadIdx.x;
  const int w = tid >> 5, lane = tid & 31, idx = lane & 15, hi = lane >> 4;
  const int bh = blockIdx.y;
  const _Float16* qp = Q  + (size_t)bh * SEQ * HDIM;
  const _Float16* kp = Kt + (size_t)bh * SEQ * HDIM;
  const _Float16* vp = Vt + (size_t)bh * HDIM * SEQ;
  const int q0 = blockIdx.x * 64 + w * 16;

  const unsigned kbuf_off = (unsigned)(size_t)(void*)kbuf;   // LDS byte offset
  const unsigned vbuf_off = (unsigned)(size_t)(void*)vtbuf;

  const v16h qf0 = load_a_frag(qp, HDIM, q0 + idx, 0, hi);
  const v16h qf1 = load_a_frag(qp, HDIM, q0 + idx, 32, hi);

  v8f o0 = {}, o1 = {}, o2 = {}, o3 = {};
  float mi[8], li[8];
#pragma unroll
  for (int r = 0; r < 8; ++r) { mi[r] = -1.0e30f; li[r] = 0.0f; }

  _Float16* pb = &pbuf[w * 16 * 40];

  for (int kc = 0; kc < SEQ; kc += 32) {
    // --- async TDM staging: wave 0 issues both tile DMAs, waits TENSORcnt ---
    if (w == 0) {
      // K chunk: 32 rows x 64 halves; LDS rows padded 32 DW + 4 DW (codes 4,3)
      tdm_load_2d(kbuf_off, &kp[(size_t)kc * HDIM],
                  /*tensor_d0=*/HDIM, /*tensor_d1=*/SEQ,
                  /*tile_d0=*/HDIM, /*tile_d1=*/32,
                  /*stride0=*/HDIM, /*pad_int=*/4, /*pad_amt=*/3);
      // Vt chunk: 64 rows x 32 halves; LDS rows padded 16 DW + 4 DW (codes 3,3)
      tdm_load_2d(vbuf_off, &vp[kc],
                  /*tensor_d0=*/SEQ, /*tensor_d1=*/HDIM,
                  /*tile_d0=*/32, /*tile_d1=*/HDIM,
                  /*stride0=*/SEQ, /*pad_int=*/3, /*pad_amt=*/3);
      __builtin_amdgcn_s_wait_tensorcnt(0);
    }
    __syncthreads();

    // scores: S[16 x 32] = Q[16 x 64] * K^T, as two 16x16 f32 tiles
    v8f s0 = {}, s1 = {};
    s0 = wmma_f16(qf0, load_b_frag(kbuf, 72, idx,      0,  hi), s0);
    s0 = wmma_f16(qf1, load_b_frag(kbuf, 72, idx,      32, hi), s0);
    s1 = wmma_f16(qf0, load_b_frag(kbuf, 72, 16 + idx, 0,  hi), s1);
    s1 = wmma_f16(qf1, load_b_frag(kbuf, 72, 16 + idx, 32, hi), s1);

    // online softmax; row of reg r is (r + 8*hi), its 16 cols live across a half
#pragma unroll
    for (int r = 0; r < 8; ++r) {
      const float a = s0[r] * 0.125f;
      const float b = s1[r] * 0.125f;
      const float rm = rmax16(fmaxf(a, b));
      const float mn = fmaxf(mi[r], rm);
      const float sc = __expf(mi[r] - mn);
      const float e0 = __expf(a - mn);
      const float e1 = __expf(b - mn);
      const float rs = rsum16(e0 + e1);
      li[r] = li[r] * sc + rs;
      mi[r] = mn;
      o0[r] *= sc; o1[r] *= sc; o2[r] *= sc; o3[r] *= sc;
      pb[(r + 8 * hi) * 40 + idx]      = (_Float16)e0;
      pb[(r + 8 * hi) * 40 + 16 + idx] = (_Float16)e1;
    }
    asm volatile("s_wait_dscnt 0" ::: "memory");        // P store -> P reload

    // O[16 x 64] += P[16 x 32] * V[32 x 64]
    const v16h pf = load_a_frag(pb, 40, idx, 0, hi);
    o0 = wmma_f16(pf, load_b_frag(vtbuf, 40, idx,      0, hi), o0);
    o1 = wmma_f16(pf, load_b_frag(vtbuf, 40, 16 + idx, 0, hi), o1);
    o2 = wmma_f16(pf, load_b_frag(vtbuf, 40, 32 + idx, 0, hi), o2);
    o3 = wmma_f16(pf, load_b_frag(vtbuf, 40, 48 + idx, 0, hi), o3);
    __syncthreads();                                    // LDS reuse next chunk
  }

  // write [B, T, C] f16 (head-interleaved) for the out-projection GEMM
  const int b = bh >> 4, head = bh & 15;
#pragma unroll
  for (int r = 0; r < 8; ++r) {
    const float inv = 1.0f / li[r];
    const int tok = q0 + r + 8 * hi;
    const size_t rowoff = ((size_t)(b * SEQ + tok)) * CH + head * HDIM;
    O[rowoff + 0 * 16 + idx] = (_Float16)(o0[r] * inv);
    O[rowoff + 1 * 16 + idx] = (_Float16)(o1[r] * inv);
    O[rowoff + 2 * 16 + idx] = (_Float16)(o2[r] * inv);
    O[rowoff + 3 * 16 + idx] = (_Float16)(o3[r] * inv);
  }
}

// ---------------------------------------------------------------------------
// 5) Out projection + bias + residual -> f32 output
// ---------------------------------------------------------------------------
__global__ void __launch_bounds__(128) out_gemm(const _Float16* __restrict__ A,
                                                const _Float16* __restrict__ Wt,
                                                const float* __restrict__ bias,
                                                const float* __restrict__ resid,
                                                float* __restrict__ Out) {
  __shared__ __align__(16) _Float16 at[64 * 40];
  __shared__ __align__(16) _Float16 bt[64 * 40];
  const int tid = threadIdx.x;
  const int w = tid >> 5, lane = tid & 31, idx = lane & 15, hi = lane >> 4;
  const int n0 = blockIdx.x * 64, m0 = blockIdx.y * 64;
  v8f acc[4] = {};
  for (int k0 = 0; k0 < CH; k0 += 32) {
#pragma unroll
    for (int i = tid; i < 256; i += 128) {
      int flat = i * 8, r = flat >> 5, c = flat & 31;
      *(half8*)&at[r * 40 + c] = *(const half8*)&A [(size_t)(m0 + r) * CH + k0 + c];
      *(half8*)&bt[r * 40 + c] = *(const half8*)&Wt[(size_t)(n0 + r) * CH + k0 + c];
      if (k0 + 32 < CH) {
        __builtin_prefetch(&A [(size_t)(m0 + r) * CH + k0 + 32 + c], 0, 3);
        __builtin_prefetch(&Wt[(size_t)(n0 + r) * CH + k0 + 32 + c], 0, 3);
      }
    }
    __syncthreads();
    v16h af = load_a_frag(at, 40, w * 16 + idx, 0, hi);
#pragma unroll
    for (int t = 0; t < 4; ++t)
      acc[t] = wmma_f16(af, load_b_frag(bt, 40, t * 16 + idx, 0, hi), acc[t]);
    __syncthreads();
  }
#pragma unroll
  for (int t = 0; t < 4; ++t) {
    const int gcol = n0 + t * 16 + idx;
    const float bv = bias[gcol];
#pragma unroll
    for (int r = 0; r < 8; ++r) {
      const int grow = m0 + w * 16 + r + 8 * hi;
      const size_t o = (size_t)grow * CH + gcol;
      Out[o] = acc[t][r] + bv + resid[o];
    }
  }
}

// ---------------------------------------------------------------------------
// Launcher
// ---------------------------------------------------------------------------
extern "C" void kernel_launch(void* const* d_in, const int* in_sizes, int n_in,
                              void* d_out, int out_size, void* d_ws, size_t ws_size,
                              hipStream_t stream) {
  (void)in_sizes; (void)n_in; (void)out_size; (void)ws_size;
  const float* x     = (const float*)d_in[0];
  const float* gam   = (const float*)d_in[1];
  const float* bet   = (const float*)d_in[2];
  const float* wqkv  = (const float*)d_in[3];
  const float* bqkv  = (const float*)d_in[4];
  const float* wout  = (const float*)d_in[5];
  const float* bout  = (const float*)d_in[6];

  char* ws = (char*)d_ws;
  // f16 workspace layout (bytes):
  _Float16* h   = (_Float16*)(ws + 0);          // 16 MB  (reused as attn out)
  _Float16* wtq = (_Float16*)(ws + 16777216);   //  6 MB  W_qkv^T
  _Float16* wto = (_Float16*)(ws + 23068672);   //  2 MB  W_out^T
  _Float16* q   = (_Float16*)(ws + 25165824);   // 16 MB  [B,H,T,64]
  _Float16* k   = (_Float16*)(ws + 41943040);   // 16 MB  [B,H,T,64]
  _Float16* vt  = (_Float16*)(ws + 58720256);   // 16 MB  [B,H,64,T]

  ln_kernel<<<ROWS, 256, 0, stream>>>(x, gam, bet, h);
  wtrans_kernel<<<(CH * C3) / 256, 256, 0, stream>>>(wqkv, wtq, CH, C3);
  wtrans_kernel<<<(CH * CH) / 256, 256, 0, stream>>>(wout, wto, CH, CH);
  qkv_gemm<<<dim3(C3 / 64, ROWS / 64), 128, 0, stream>>>(h, wtq, bqkv, q, k, vt);
  attn_kernel<<<dim3(SEQ / 64, BATCH * HEADS), 128, 0, stream>>>(q, k, vt, h);
  out_gemm<<<dim3(CH / 64, ROWS / 64), 128, 0, stream>>>(h, wto, bout, x,
                                                         (float*)d_out);
}
